// WindowAttention_49744311222357
// MI455X (gfx1250) — matrix operand order repaired and checked by reading, not compiled, not run on hardware
//
#include <hip/hip_runtime.h>
#include <hip/hip_bf16.h>

typedef __bf16 bf16_t;
typedef __attribute__((ext_vector_type(16))) bf16_t v16bf;
typedef __attribute__((ext_vector_type(8)))  float  v8f;

static constexpr int   BATCH = 16, NTOK = 4096, DIM = 768, HEADS = 12, HD = 64;
static constexpr int   ROWS  = BATCH * NTOK;     // 65536
static constexpr int   NQKV  = 3 * DIM;          // 2304
static constexpr float SCALE = 0.125f;           // 64^-0.5

// workspace layout (bytes)
static constexpr size_t SZ_WHT    = (size_t)NQKV * DIM * 2;          // 3.5 MB
static constexpr size_t SZ_WPT    = (size_t)DIM * DIM * 2;           // 1.2 MB
static constexpr size_t SZ_WINBUF = (size_t)1024 * 12 * 64 * 64 * 2; // 100.7 MB
static constexpr size_t OFF_WHT = 0;
static constexpr size_t OFF_WPT = OFF_WHT + SZ_WHT;
static constexpr size_t OFF_QW  = OFF_WPT + SZ_WPT;
static constexpr size_t OFF_KW  = OFF_QW + SZ_WINBUF;
static constexpr size_t OFF_VT  = OFF_KW + SZ_WINBUF;
static constexpr size_t OFF_AO  = OFF_VT + SZ_WINBUF;

union BFrag {
    v16bf  v;
    uint4  q[2];
    bf16_t e[16];
};

static __device__ inline v8f wmma_bf16(v16bf a, v16bf b, v8f c) {
    return __builtin_amdgcn_wmma_f32_16x16x32_bf16(false, a, false, b, (short)0, c,
                                                   false, false);
}

// packs to a single v_cvt_pk_bf16_f32
static __device__ inline unsigned pack_bf16(float a, float b) {
    union { bf16_t h[2]; unsigned u; } x;
    x.h[0] = (bf16_t)a;
    x.h[1] = (bf16_t)b;
    return x.u;
}

// ---------------------------------------------------------------------------
// Kernel 1: coalesced LDS-tiled transpose + f32->bf16 : dst[N][K] = src[K][N]
//   grid = (N/32, K/32), block = (32,8)
// ---------------------------------------------------------------------------
__global__ __launch_bounds__(256) void transpose_bf16_kernel(
    const float* __restrict__ src, bf16_t* __restrict__ dst, int K, int N)
{
    __shared__ float t[32][33];
    const int n0 = blockIdx.x * 32, k0 = blockIdx.y * 32;
    const int tx = threadIdx.x, ty = threadIdx.y;
    #pragma unroll
    for (int i = 0; i < 32; i += 8)
        t[ty + i][tx] = src[(size_t)(k0 + ty + i) * N + n0 + tx];
    __syncthreads();
    #pragma unroll
    for (int i = 0; i < 32; i += 8)
        dst[(size_t)(n0 + ty + i) * K + k0 + tx] = (bf16_t)t[tx][ty + i];
}

// ---------------------------------------------------------------------------
// Kernel 2: QKV GEMM (bf16 WMMA) + bias + window partition + 2D RoPE + scale
//   grid = (NQKV/128, ROWS/64), block = 64 (2 waves, each 64x64 output tile)
//   Q,K -> [win][head][tok][hd] bf16 ; V -> [win][head][hd][tok] bf16
// ---------------------------------------------------------------------------
__global__ __launch_bounds__(64) void qkv_rope_kernel(
    const float* __restrict__ x, const bf16_t* __restrict__ whT,
    const float* __restrict__ bqkv,
    bf16_t* __restrict__ Qw, bf16_t* __restrict__ Kw, bf16_t* __restrict__ VwT)
{
    __shared__ __align__(16) float sC[2][64 * 65];
    const int lane = threadIdx.x & 31;
    const int wave = threadIdx.x >> 5;
    const int l16  = lane & 15;
    const int hi   = (lane >> 4) & 1;
    const int Mbase = blockIdx.y * 64;
    const int Nbase = blockIdx.x * 128 + wave * 64;

    v8f acc[4][4];
    #pragma unroll
    for (int i = 0; i < 4; i++)
        #pragma unroll
        for (int j = 0; j < 4; j++)
            #pragma unroll
            for (int e = 0; e < 8; e++) acc[i][j][e] = 0.f;

    for (int k0 = 0; k0 < DIM; k0 += 32) {
        BFrag A[4], Bm[4];
        #pragma unroll
        for (int tm = 0; tm < 4; tm++) {
            // A lane layout: lanes 0-15 rows, K chunks k0..k0+7 & k0+16..23;
            // lanes 16-31: +8 on each chunk.
            const float* xr = x + (size_t)(Mbase + tm * 16 + l16) * DIM + k0 + hi * 8;
            float4 f0 = *(const float4*)(xr);
            float4 f1 = *(const float4*)(xr + 4);
            float4 f2 = *(const float4*)(xr + 16);
            float4 f3 = *(const float4*)(xr + 20);
            A[tm].e[0]  = (bf16_t)f0.x; A[tm].e[1]  = (bf16_t)f0.y;
            A[tm].e[2]  = (bf16_t)f0.z; A[tm].e[3]  = (bf16_t)f0.w;
            A[tm].e[4]  = (bf16_t)f1.x; A[tm].e[5]  = (bf16_t)f1.y;
            A[tm].e[6]  = (bf16_t)f1.z; A[tm].e[7]  = (bf16_t)f1.w;
            A[tm].e[8]  = (bf16_t)f2.x; A[tm].e[9]  = (bf16_t)f2.y;
            A[tm].e[10] = (bf16_t)f2.z; A[tm].e[11] = (bf16_t)f2.w;
            A[tm].e[12] = (bf16_t)f3.x; A[tm].e[13] = (bf16_t)f3.y;
            A[tm].e[14] = (bf16_t)f3.z; A[tm].e[15] = (bf16_t)f3.w;
        }
        #pragma unroll
        for (int tn = 0; tn < 4; tn++) {
            // B lane layout: lane = column, K chunks k0+hi*8..+7 & k0+16+hi*8..+7.
            const bf16_t* wr = whT + (size_t)(Nbase + tn * 16 + l16) * DIM + k0 + hi * 8;
            Bm[tn].q[0] = *(const uint4*)(wr);
            Bm[tn].q[1] = *(const uint4*)(wr + 16);
        }
        #pragma unroll
        for (int tm = 0; tm < 4; tm++)
            #pragma unroll
            for (int tn = 0; tn < 4; tn++)
                acc[tm][tn] = wmma_bf16(A[tm].v, Bm[tn].v, acc[tm][tn]);
    }

    // stage C (+bias) to LDS, stride 65 to kill bank conflicts
    float* sc = sC[wave];
    #pragma unroll
    for (int tm = 0; tm < 4; tm++) {
        int mrow = tm * 16 + hi * 8;
        #pragma unroll
        for (int tn = 0; tn < 4; tn++) {
            int ncol = tn * 16 + l16;
            float bb = bqkv[Nbase + ncol];
            #pragma unroll
            for (int i = 0; i < 8; i++)
                sc[(mrow + i) * 65 + ncol] = acc[tm][tn][i] + bb;
        }
    }
    __syncthreads();

    // epilogue: this 64-col tile is one (q/k/v, head) pair; the 64 M rows share
    // one grid row h (Mbase is 64-aligned), w = row.
    const int t3   = Nbase / DIM;          // 0=q,1=k,2=v
    const int head = (Nbase % DIM) / HD;
    const int b    = Mbase >> 12;
    const int h    = (Mbase & 4095) >> 6;
    const int winbase = (b << 6) | ((h >> 3) << 3);
    const int posh = h & 7;

    // inv_freq[j] = 10000^(-j/16); literal table folds under full unroll
    const float INVF[16] = {
        1.0f,                    0.5623413251903491f,    0.31622776601683794f,
        0.17782794100389226f,    0.1f,                   0.05623413251903491f,
        0.031622776601683794f,   0.017782794100389224f,  0.01f,
        0.005623413251903491f,   0.0031622776601683794f, 0.0017782794100389228f,
        0.001f,                  0.0005623413251903491f, 0.00031622776601683794f,
        0.00017782794100389227f };

    if (t3 < 2) {
        bf16_t* dst = (t3 == 0) ? Qw : Kw;
        const float sca = (t3 == 0) ? SCALE : 1.0f;
        #pragma unroll
        for (int rr = 0; rr < 2; rr++) {
            int row   = lane * 2 + rr;               // == w
            int win   = winbase | (row >> 3);
            int intok = (posh << 3) | (row & 7);
            float posw = (float)(row & 7);
            bf16_t* drow = dst + ((((size_t)win * HEADS + head) * 64 + intok) << 6);
            #pragma unroll
            for (int p = 0; p < 32; p++) {
                float v1 = sc[row * 65 + 2 * p];
                float v2 = sc[row * 65 + 2 * p + 1];
                int   j   = (p < 16) ? p : (p - 16);
                float pos = (p < 16) ? (float)posh : posw;
                float ang = pos * INVF[j];
                float sn, cn;
                __sincosf(ang, &sn, &cn);
                *(unsigned*)(drow + 2 * p) =
                    pack_bf16((v1 * cn - v2 * sn) * sca, (v1 * sn + v2 * cn) * sca);
            }
        }
    } else {
        // V transposed: [win][head][hd][tok]; rows wb*8..wb*8+7 -> contiguous intok
        #pragma unroll
        for (int cc = 0; cc < 2; cc++) {
            int c = lane * 2 + cc;                   // hd
            for (int wb = 0; wb < 8; wb++) {
                int win = winbase | wb;
                bf16_t* dcol = VwT + ((((size_t)win * HEADS + head) * 64 + c) << 6)
                             + (posh << 3);
                #pragma unroll
                for (int q2 = 0; q2 < 4; q2++) {
                    int row = wb * 8 + q2 * 2;
                    *(unsigned*)(dcol + q2 * 2) =
                        pack_bf16(sc[row * 65 + c], sc[(row + 1) * 65 + c]);
                }
            }
        }
    }
}

// ---------------------------------------------------------------------------
// Kernel 3: windowed attention. One wave per (window, head).
//   S = Q K^T (scale folded into Q), softmax, O = P V, write merged layout bf16
//   grid = 12288/2, block = 64 (2 waves)
// ---------------------------------------------------------------------------
__global__ __launch_bounds__(64) void attn_kernel(
    const bf16_t* __restrict__ Qw, const bf16_t* __restrict__ Kw,
    const bf16_t* __restrict__ VwT, bf16_t* __restrict__ Aout)
{
    __shared__ __align__(16) float          sS[2][64 * 65];
    __shared__ __align__(16) unsigned short sP[2][64 * 64];
    const int lane = threadIdx.x & 31;
    const int wave = threadIdx.x >> 5;
    const int l16  = lane & 15;
    const int hi   = (lane >> 4) & 1;
    const int wh   = blockIdx.x * 2 + wave;          // win*HEADS + head
    const int win  = wh / HEADS;
    const int head = wh - win * HEADS;
    const bf16_t* Qb = Qw  + ((size_t)wh << 12);
    const bf16_t* Kb = Kw  + ((size_t)wh << 12);
    const bf16_t* Vb = VwT + ((size_t)wh << 12);

    // ---- S = Q @ K^T -------------------------------------------------------
    v8f acc[4][4];
    #pragma unroll
    for (int i = 0; i < 4; i++)
        #pragma unroll
        for (int j = 0; j < 4; j++)
            #pragma unroll
            for (int e = 0; e < 8; e++) acc[i][j][e] = 0.f;

    #pragma unroll
    for (int k0 = 0; k0 < 64; k0 += 32) {
        BFrag A[4], Bm[4];
        #pragma unroll
        for (int tm = 0; tm < 4; tm++) {
            const bf16_t* qa = Qb + (size_t)(tm * 16 + l16) * 64 + k0 + hi * 8;
            A[tm].q[0] = *(const uint4*)(qa);
            A[tm].q[1] = *(const uint4*)(qa + 16);
        }
        #pragma unroll
        for (int tn = 0; tn < 4; tn++) {
            const bf16_t* kb = Kb + (size_t)(tn * 16 + l16) * 64 + k0 + hi * 8;
            Bm[tn].q[0] = *(const uint4*)(kb);
            Bm[tn].q[1] = *(const uint4*)(kb + 16);
        }
        #pragma unroll
        for (int tm = 0; tm < 4; tm++)
            #pragma unroll
            for (int tn = 0; tn < 4; tn++)
                acc[tm][tn] = wmma_bf16(A[tm].v, Bm[tn].v, acc[tm][tn]);
    }

    float* sc = sS[wave];
    #pragma unroll
    for (int tm = 0; tm < 4; tm++) {
        int mrow = tm * 16 + hi * 8;
        #pragma unroll
        for (int tn = 0; tn < 4; tn++) {
            int ncol = tn * 16 + l16;
            #pragma unroll
            for (int i = 0; i < 8; i++)
                sc[(mrow + i) * 65 + ncol] = acc[tm][tn][i];
        }
    }
    __syncthreads();

    // ---- softmax over rows, pack P as bf16 into 16B-aligned LDS ------------
    unsigned short* pp = sP[wave];
    #pragma unroll
    for (int rr = 0; rr < 2; rr++) {
        int row = lane * 2 + rr;
        float mx = -1e30f;
        for (int c = 0; c < 64; c++) mx = fmaxf(mx, sc[row * 65 + c]);
        float sum = 0.f;
        for (int c = 0; c < 64; c++) {
            float t = __expf(sc[row * 65 + c] - mx);
            sum += t;
            sc[row * 65 + c] = t;
        }
        float r = 1.0f / sum;
        for (int c = 0; c < 64; c += 2)
            *(unsigned*)&pp[row * 64 + c] =
                pack_bf16(sc[row * 65 + c] * r, sc[row * 65 + c + 1] * r);
    }
    __syncthreads();

    // ---- O = P @ V ---------------------------------------------------------
    v8f o[4][4];
    #pragma unroll
    for (int i = 0; i < 4; i++)
        #pragma unroll
        for (int j = 0; j < 4; j++)
            #pragma unroll
            for (int e = 0; e < 8; e++) o[i][j][e] = 0.f;

    #pragma unroll
    for (int k0 = 0; k0 < 64; k0 += 32) {
        BFrag A[4], Bm[4];
        #pragma unroll
        for (int tm = 0; tm < 4; tm++) {
            const unsigned short* pa = pp + (tm * 16 + l16) * 64 + k0 + hi * 8;
            A[tm].q[0] = *(const uint4*)(pa);
            A[tm].q[1] = *(const uint4*)(pa + 16);
        }
        #pragma unroll
        for (int tn = 0; tn < 4; tn++) {
            const bf16_t* vb = Vb + (size_t)(tn * 16 + l16) * 64 + k0 + hi * 8;
            Bm[tn].q[0] = *(const uint4*)(vb);
            Bm[tn].q[1] = *(const uint4*)(vb + 16);
        }
        #pragma unroll
        for (int tm = 0; tm < 4; tm++)
            #pragma unroll
            for (int tn = 0; tn < 4; tn++)
                o[tm][tn] = wmma_bf16(A[tm].v, Bm[tn].v, o[tm][tn]);
    }

    #pragma unroll
    for (int tm = 0; tm < 4; tm++) {
        int mrow = tm * 16 + hi * 8;
        #pragma unroll
        for (int tn = 0; tn < 4; tn++) {
            int ncol = tn * 16 + l16;
            #pragma unroll
            for (int i = 0; i < 8; i++)
                sc[(mrow + i) * 65 + ncol] = o[tm][tn][i];
        }
    }
    __syncthreads();

    // ---- merge back to (B, N, DIM) bf16 ------------------------------------
    const int b   = win >> 6;
    const int wwh = (win >> 3) & 7;
    const int www = win & 7;
    #pragma unroll
    for (int rr = 0; rr < 2; rr++) {
        int tk = lane * 2 + rr;                      // in-window token
        int h  = wwh * 8 + (tk >> 3);
        int w  = www * 8 + (tk & 7);
        size_t grow = (size_t)b * NTOK + (size_t)h * 64 + w;
        bf16_t* dr = Aout + grow * DIM + head * HD;
        #pragma unroll
        for (int p = 0; p < 32; p++)
            *(unsigned*)(dr + 2 * p) =
                pack_bf16(sc[tk * 65 + 2 * p], sc[tk * 65 + 2 * p + 1]);
    }
}

// ---------------------------------------------------------------------------
// Kernel 4: output projection: out(f32) = Aout(bf16) @ w_proj + b_proj
//   grid = (DIM/128, ROWS/64), block = 64 (2 waves, each 64x64)
// ---------------------------------------------------------------------------
__global__ __launch_bounds__(64) void proj_kernel(
    const bf16_t* __restrict__ Aout, const bf16_t* __restrict__ wpT,
    const float* __restrict__ bproj, float* __restrict__ out)
{
    const int lane = threadIdx.x & 31;
    const int wave = threadIdx.x >> 5;
    const int l16  = lane & 15;
    const int hi   = (lane >> 4) & 1;
    const int Mbase = blockIdx.y * 64;
    const int Nbase = blockIdx.x * 128 + wave * 64;

    v8f acc[4][4];
    #pragma unroll
    for (int i = 0; i < 4; i++)
        #pragma unroll
        for (int j = 0; j < 4; j++)
            #pragma unroll
            for (int e = 0; e < 8; e++) acc[i][j][e] = 0.f;

    for (int k0 = 0; k0 < DIM; k0 += 32) {
        BFrag A[4], Bm[4];
        #pragma unroll
        for (int tm = 0; tm < 4; tm++) {
            const bf16_t* ar = Aout + (size_t)(Mbase + tm * 16 + l16) * DIM + k0 + hi * 8;
            A[tm].q[0] = *(const uint4*)(ar);
            A[tm].q[1] = *(const uint4*)(ar + 16);
        }
        #pragma unroll
        for (int tn = 0; tn < 4; tn++) {
            const bf16_t* wr = wpT + (size_t)(Nbase + tn * 16 + l16) * DIM + k0 + hi * 8;
            Bm[tn].q[0] = *(const uint4*)(wr);
            Bm[tn].q[1] = *(const uint4*)(wr + 16);
        }
        #pragma unroll
        for (int tm = 0; tm < 4; tm++)
            #pragma unroll
            for (int tn = 0; tn < 4; tn++)
                acc[tm][tn] = wmma_bf16(A[tm].v, Bm[tn].v, acc[tm][tn]);
    }

    #pragma unroll
    for (int tm = 0; tm < 4; tm++) {
        int mrow = Mbase + tm * 16 + hi * 8;
        #pragma unroll
        for (int tn = 0; tn < 4; tn++) {
            int col = Nbase + tn * 16 + l16;
            float bb = bproj[col];
            #pragma unroll
            for (int i = 0; i < 8; i++)
                out[(size_t)(mrow + i) * DIM + col] = acc[tm][tn][i] + bb;
        }
    }
}

// ---------------------------------------------------------------------------
extern "C" void kernel_launch(void* const* d_in, const int* in_sizes, int n_in,
                              void* d_out, int out_size, void* d_ws, size_t ws_size,
                              hipStream_t stream)
{
    (void)in_sizes; (void)n_in; (void)out_size; (void)ws_size;
    const float* x     = (const float*)d_in[0];
    const float* wqkv  = (const float*)d_in[1];
    const float* bqkv  = (const float*)d_in[2];
    const float* wproj = (const float*)d_in[3];
    const float* bproj = (const float*)d_in[4];
    float* out = (float*)d_out;

    char* ws = (char*)d_ws;
    bf16_t* whT = (bf16_t*)(ws + OFF_WHT);
    bf16_t* wpT = (bf16_t*)(ws + OFF_WPT);
    bf16_t* Qw  = (bf16_t*)(ws + OFF_QW);
    bf16_t* Kw  = (bf16_t*)(ws + OFF_KW);
    bf16_t* VwT = (bf16_t*)(ws + OFF_VT);
    bf16_t* Ao  = (bf16_t*)(ws + OFF_AO);

    {
        dim3 grid(NQKV / 32, DIM / 32), blk(32, 8);
        transpose_bf16_kernel<<<grid, blk, 0, stream>>>(wqkv, whT, DIM, NQKV);
    }
    {
        dim3 grid(DIM / 32, DIM / 32), blk(32, 8);
        transpose_bf16_kernel<<<grid, blk, 0, stream>>>(wproj, wpT, DIM, DIM);
    }
    {
        dim3 grid(NQKV / 128, ROWS / 64);
        qkv_rope_kernel<<<grid, 64, 0, stream>>>(x, whT, bqkv, Qw, Kw, VwT);
    }
    {
        int wh_total = 1024 * HEADS;   // 12288 (window, head) pairs, 2 per block
        attn_kernel<<<wh_total / 2, 64, 0, stream>>>(Qw, Kw, VwT, Ao);
    }
    {
        dim3 grid(DIM / 128, ROWS / 64);
        proj_kernel<<<grid, 64, 0, stream>>>(Ao, wpT, bproj, out);
    }
}
